// CausalSelfAttention_76888504533156
// MI455X (gfx1250) — matrix operand order repaired and checked by reading, not compiled, not run on hardware
//
#include <hip/hip_runtime.h>

typedef __attribute__((ext_vector_type(16))) _Float16 v16h;
typedef __attribute__((ext_vector_type(8)))  float    v8f;
typedef _Float16 half_t;

#define T_SEQ 2048
#define DIMC  2048
#define NH    16
#define HD    128

union Frag16 { v16h v; uint4 u[2]; };

// A fragment: rows m0..m0+15, K columns k0..k0+31, src row-major with leading dim ld.
// Lane l: row m0 + l%16, halves at k0+(l/16)*8 (u[0]) and k0+16+(l/16)*8 (u[1]).
__device__ __forceinline__ void load_frag_a(Frag16& f, const half_t* __restrict__ src,
                                            int ld, int m0, int k0) {
  const int lane = threadIdx.x & 31;
  const half_t* p = src + (size_t)(m0 + (lane & 15)) * ld + k0 + ((lane >> 4) << 3);
  const uint4* q = reinterpret_cast<const uint4*>(p);
  f.u[0] = q[0];   // K = +0..7
  f.u[1] = q[2];   // K = +16..23
}

// B fragment: cols n0..n0+15, K rows k0..k0+31, src stored as B^T row-major (row n, contiguous K).
// Lane l: col n0 + l%16, halves at k0+(l/16)*8 (u[0]) and k0+16+(l/16)*8 (u[1]).
__device__ __forceinline__ void load_frag_b(Frag16& f, const half_t* __restrict__ src,
                                            int ld, int n0, int k0) {
  const int lane = threadIdx.x & 31;
  const half_t* p = src + (size_t)(n0 + (lane & 15)) * ld + k0 + ((lane >> 4) << 3);
  const uint4* q = reinterpret_cast<const uint4*>(p);
  f.u[0] = q[0];   // K = +0..7
  f.u[1] = q[2];   // K = +16..23
}

__device__ __forceinline__ v8f wmma_f16(const Frag16& a, const Frag16& b, v8f c) {
  return __builtin_amdgcn_wmma_f32_16x16x32_f16(false, a.v, false, b.v, (short)0, c, false, false);
}

// ---------------------------------------------------------------------------
// f32 -> f16 conversion
// ---------------------------------------------------------------------------
__global__ void cvt_f32_f16(const float* __restrict__ in, half_t* __restrict__ out, int n) {
  int i = (blockIdx.x * blockDim.x + threadIdx.x) * 4;
  if (i < n) {
    float4 v = *(const float4*)(in + i);
    out[i + 0] = (half_t)v.x;
    out[i + 1] = (half_t)v.y;
    out[i + 2] = (half_t)v.z;
    out[i + 3] = (half_t)v.w;
  }
}

// ---------------------------------------------------------------------------
// C(MxN,f32) = A(MxK,f16 row-major) * B^T, with B given as (N x K) row-major.
// One wave computes a 64x64 tile (4x4 WMMA accumulators); 8 waves / block.
// kk += 32 loop: the compiler unrolls and software-pipelines it with staged
// partial s_wait_loadcnt waits (verified in Round-1 disassembly).
// ---------------------------------------------------------------------------
__global__ __launch_bounds__(256) void gemm_ab_t(const half_t* __restrict__ A,
                                                 const half_t* __restrict__ B,
                                                 float* __restrict__ C,
                                                 int M, int N, int K) {
  const int lane = threadIdx.x & 31;
  const int wave = threadIdx.x >> 5;
  const int tiles_n = N >> 6;
  const int tid = blockIdx.x * 8 + wave;
  const int m0 = (tid / tiles_n) << 6;
  const int n0 = (tid % tiles_n) << 6;

  v8f acc[4][4];
  #pragma unroll
  for (int i = 0; i < 4; ++i)
    #pragma unroll
    for (int j = 0; j < 4; ++j)
      acc[i][j] = (v8f){};

  for (int kk = 0; kk < K; kk += 32) {
    Frag16 a[4], b[4];
    #pragma unroll
    for (int i = 0; i < 4; ++i) load_frag_a(a[i], A, K, m0 + 16 * i, kk);
    #pragma unroll
    for (int j = 0; j < 4; ++j) load_frag_b(b[j], B, K, n0 + 16 * j, kk);
    #pragma unroll
    for (int i = 0; i < 4; ++i)
      #pragma unroll
      for (int j = 0; j < 4; ++j)
        acc[i][j] = wmma_f16(a[i], b[j], acc[i][j]);
  }

  const int rbase = m0 + ((lane >> 4) << 3);
  const int cbase = n0 + (lane & 15);
  #pragma unroll
  for (int i = 0; i < 4; ++i)
    #pragma unroll
    for (int j = 0; j < 4; ++j)
      #pragma unroll
      for (int v = 0; v < 8; ++v)
        C[(size_t)(rbase + 16 * i + v) * N + (cbase + 16 * j)] = acc[i][j][v];
}

// ---------------------------------------------------------------------------
// Per-(t,h) epilogue: RMS-norm q/k, RoPE, v = l0*v + l1*ve.
// Writes qh/kh as [h][t][d] f16 and v transposed as vth [h][d][t] f16.
// One wave per (t,h) row; lane owns 4 consecutive d.
// ---------------------------------------------------------------------------
__global__ __launch_bounds__(256) void qkv_epilogue(const float* __restrict__ qkv,
                                                    const float* __restrict__ ve,
                                                    const float* __restrict__ lambdas,
                                                    half_t* __restrict__ qh,
                                                    half_t* __restrict__ kh,
                                                    half_t* __restrict__ vth) {
  const int lane = threadIdx.x & 31;
  const int row = blockIdx.x * 8 + (threadIdx.x >> 5);
  const int t = row >> 4;
  const int h = row & 15;
  const int d0 = lane * 4;

  const float* base = qkv + (size_t)t * (3 * DIMC) + h * HD;
  float4 q4 = *(const float4*)(base + d0);
  float4 k4 = *(const float4*)(base + DIMC + d0);
  float4 v4 = *(const float4*)(base + 2 * DIMC + d0);

  float sq = q4.x * q4.x + q4.y * q4.y + q4.z * q4.z + q4.w * q4.w;
  float sk = k4.x * k4.x + k4.y * k4.y + k4.z * k4.z + k4.w * k4.w;
  #pragma unroll
  for (int m = 1; m < 32; m <<= 1) { sq += __shfl_xor(sq, m); sk += __shfl_xor(sk, m); }
  const float eps = 1.1920929e-07f;
  const float rq = rsqrtf(sq * (1.0f / HD) + eps);
  const float rk = rsqrtf(sk * (1.0f / HD) + eps);

  float qv[4] = {q4.x * rq, q4.y * rq, q4.z * rq, q4.w * rq};
  float kv[4] = {k4.x * rk, k4.y * rk, k4.z * rk, k4.w * rk};

  const bool hi = lane >= 16;  // upper half holds x2 (d in 64..127)
  half_t* qdst = qh + ((size_t)h * T_SEQ + t) * HD + d0;
  half_t* kdst = kh + ((size_t)h * T_SEQ + t) * HD + d0;
  #pragma unroll
  for (int j = 0; j < 4; ++j) {
    const int dl = (d0 + j) & 63;
    const float freq = (dl < 32) ? exp2f((float)dl * (-10.0f / 31.0f)) : 0.0f;
    float s, c;
    sincosf((float)t * freq, &s, &c);
    const float qp = __shfl_xor(qv[j], 16);
    const float kp = __shfl_xor(kv[j], 16);
    const float qy = hi ? (qv[j] * c - qp * s) : (qv[j] * c + qp * s);
    const float ky = hi ? (kv[j] * c - kp * s) : (kv[j] * c + kp * s);
    qdst[j] = (half_t)qy;
    kdst[j] = (half_t)ky;
  }

  const float l0 = lambdas[0], l1 = lambdas[1];
  float4 e4 = *(const float4*)(ve + ((size_t)t * NH + h) * HD + d0);
  float vm[4] = {l0 * v4.x + l1 * e4.x, l0 * v4.y + l1 * e4.y,
                 l0 * v4.z + l1 * e4.z, l0 * v4.w + l1 * e4.w};
  #pragma unroll
  for (int j = 0; j < 4; ++j)
    vth[((size_t)h * HD + d0 + j) * T_SEQ + t] = (half_t)vm[j];
}

// ---------------------------------------------------------------------------
// Flash attention: one wave = 16 queries of one head; 32-key chunks,
// online softmax, causal mask analytic (block_mask input unused).
// P goes through per-wave LDS tile (C-layout f32 -> A-layout f16).
// ---------------------------------------------------------------------------
__global__ __launch_bounds__(256) void flash_attn(const half_t* __restrict__ qh,
                                                  const half_t* __restrict__ kh,
                                                  const half_t* __restrict__ vth,
                                                  half_t* __restrict__ yh) {
  const int lane = threadIdx.x & 31;
  const int wave = threadIdx.x >> 5;
  const int wid = blockIdx.x * 8 + wave;
  const int qb = wid & 127;   // T/16 = 128 query blocks per head
  const int h = wid >> 7;
  const int q0 = qb << 4;

  __shared__ half_t plds[8][16 * 32];
  half_t* pw = plds[wave];

  const half_t* qbase = qh + (size_t)h * T_SEQ * HD;
  const half_t* kbase = kh + (size_t)h * T_SEQ * HD;
  const half_t* vbase = vth + (size_t)h * HD * T_SEQ;

  Frag16 qf[4];
  #pragma unroll
  for (int d = 0; d < 4; ++d) load_frag_a(qf[d], qbase, HD, q0, d * 32);

  float mr[8], lr[8];
  v8f o[8];
  #pragma unroll
  for (int v = 0; v < 8; ++v) { mr[v] = -3.0e38f; lr[v] = 0.0f; }
  #pragma unroll
  for (int f = 0; f < 8; ++f) o[f] = (v8f){};

  const float scale = 0.08838834764831845f;  // D^-0.5
  const int kend = q0 + 16;

  for (int kk = 0; kk < kend; kk += 32) {
    // S = Q * K^T for this 16x32 chunk
    v8f s0 = (v8f){}, s1 = (v8f){};
    #pragma unroll
    for (int d = 0; d < 4; ++d) {
      Frag16 kf0, kf1;
      load_frag_b(kf0, kbase, HD, kk, d * 32);
      load_frag_b(kf1, kbase, HD, kk + 16, d * 32);
      s0 = wmma_f16(qf[d], kf0, s0);
      s1 = wmma_f16(qf[d], kf1, s1);
    }

    // scale + causal mask + row max (rows live in vgpr index v, half-wave)
    const int qrow0 = q0 + ((lane >> 4) << 3);
    const int c0 = kk + (lane & 15);
    float rm[8];
    #pragma unroll
    for (int v = 0; v < 8; ++v) {
      const int q = qrow0 + v;
      float a = s0[v] * scale; if (c0 > q)      a = -1.0e30f;
      float b = s1[v] * scale; if (c0 + 16 > q) b = -1.0e30f;
      s0[v] = a; s1[v] = b;
      rm[v] = fmaxf(a, b);
    }
    #pragma unroll
    for (int v = 0; v < 8; ++v)
      #pragma unroll
      for (int m = 1; m < 16; m <<= 1) rm[v] = fmaxf(rm[v], __shfl_xor(rm[v], m));

    float alpha[8], rs[8];
    #pragma unroll
    for (int v = 0; v < 8; ++v) {
      const float mn = fmaxf(mr[v], rm[v]);
      alpha[v] = __expf(mr[v] - mn);
      mr[v] = mn;
      const float p0 = __expf(s0[v] - mn);
      const float p1 = __expf(s1[v] - mn);
      s0[v] = p0; s1[v] = p1;
      rs[v] = p0 + p1;
    }
    #pragma unroll
    for (int v = 0; v < 8; ++v) {
      #pragma unroll
      for (int m = 1; m < 16; m <<= 1) rs[v] += __shfl_xor(rs[v], m);
      lr[v] = lr[v] * alpha[v] + rs[v];
    }

    // P: C-layout f32 -> LDS f16 (per-wave tile; DS ops are in-order per wave)
    #pragma unroll
    for (int v = 0; v < 8; ++v) {
      const int r = ((lane >> 4) << 3) + v;
      pw[r * 32 + (lane & 15)] = (half_t)s0[v];
      pw[r * 32 + 16 + (lane & 15)] = (half_t)s1[v];
    }
    __asm__ volatile("" ::: "memory");

    Frag16 pf;
    {
      const uint4* lp = reinterpret_cast<const uint4*>(pw + (lane & 15) * 32 + ((lane >> 4) << 3));
      pf.u[0] = lp[0];  // K = +0..7
      pf.u[1] = lp[2];  // K = +16..23
    }

    // O = O*alpha + P * V_chunk
    #pragma unroll
    for (int f = 0; f < 8; ++f) {
      #pragma unroll
      for (int v = 0; v < 8; ++v) o[f][v] *= alpha[v];
      Frag16 vf;
      load_frag_b(vf, vbase, T_SEQ, f * 16, kk);
      o[f] = wmma_f16(pf, vf, o[f]);
    }
  }

  float inv[8];
  #pragma unroll
  for (int v = 0; v < 8; ++v) inv[v] = 1.0f / lr[v];
  #pragma unroll
  for (int f = 0; f < 8; ++f)
    #pragma unroll
    for (int v = 0; v < 8; ++v) {
      const int r = ((lane >> 4) << 3) + v;
      yh[(size_t)(q0 + r) * DIMC + h * HD + f * 16 + (lane & 15)] = (half_t)(o[f][v] * inv[v]);
    }
}

// ---------------------------------------------------------------------------
extern "C" void kernel_launch(void* const* d_in, const int* in_sizes, int n_in,
                              void* d_out, int out_size, void* d_ws, size_t ws_size,
                              hipStream_t stream) {
  const float* x        = (const float*)d_in[0];
  const float* ve       = (const float*)d_in[1];
  // d_in[2] = block_mask (causal mask computed analytically; unused)
  const float* qkv_w    = (const float*)d_in[3];
  const float* lambdas  = (const float*)d_in[4];
  const float* c_proj_w = (const float*)d_in[5];
  float* out = (float*)d_out;

  char* ws = (char*)d_ws;
  size_t off = 0;
  half_t* xh   = (half_t*)(ws + off); off += (size_t)T_SEQ * DIMC * 2;      // x f16
  half_t* whq  = (half_t*)(ws + off); off += (size_t)3 * DIMC * DIMC * 2;   // qkv_w f16
  half_t* whp  = (half_t*)(ws + off); off += (size_t)DIMC * DIMC * 2;       // c_proj_w f16
  float*  qkvf = (float*)(ws + off);  off += (size_t)T_SEQ * 3 * DIMC * 4;  // qkv f32
  half_t* qh   = (half_t*)(ws + off); off += (size_t)NH * T_SEQ * HD * 2;   // q [h][t][d]
  half_t* kh   = (half_t*)(ws + off); off += (size_t)NH * T_SEQ * HD * 2;   // k [h][t][d]
  half_t* vth  = (half_t*)(ws + off); off += (size_t)NH * HD * T_SEQ * 2;   // v [h][d][t]
  half_t* yh   = (half_t*)(ws + off); off += (size_t)T_SEQ * DIMC * 2;      // y f16

  const int nx = T_SEQ * DIMC;
  const int nw = 3 * DIMC * DIMC;
  const int np = DIMC * DIMC;
  cvt_f32_f16<<<(nx / 4 + 255) / 256, 256, 0, stream>>>(x, xh, nx);
  cvt_f32_f16<<<(nw / 4 + 255) / 256, 256, 0, stream>>>(qkv_w, whq, nw);
  cvt_f32_f16<<<(np / 4 + 255) / 256, 256, 0, stream>>>(c_proj_w, whp, np);

  // qkv = x @ qkv_w^T : (2048 x 6144), tiles 32x96 = 3072 waves / 8 = 384 blocks
  gemm_ab_t<<<(T_SEQ / 64) * (3 * DIMC / 64) / 8, 256, 0, stream>>>(xh, whq, qkvf,
                                                                    T_SEQ, 3 * DIMC, DIMC);
  // RMS-norm + RoPE + v mix
  qkv_epilogue<<<T_SEQ * NH / 8, 256, 0, stream>>>(qkvf, ve, lambdas, qh, kh, vth);
  // attention: 16 heads * 128 query blocks = 2048 waves / 8 = 256 blocks
  flash_attn<<<NH * (T_SEQ / 16) / 8, 256, 0, stream>>>(qh, kh, vth, yh);
  // out = y @ c_proj_w^T : (2048 x 2048), 1024 waves / 8 = 128 blocks
  gemm_ab_t<<<(T_SEQ / 64) * (DIMC / 64) / 8, 256, 0, stream>>>(yh, whp, out,
                                                                T_SEQ, DIMC, DIMC);
}
